// EfficientAttention_65867618452183
// MI455X (gfx1250) — matrix operand order repaired and checked by reading, not compiled, not run on hardware
//
#include <hip/hip_runtime.h>
#include <hip/hip_bf16.h>

// ---------------------------------------------------------------------------
// EfficientAttention (pruned cosFormer) for MI455X / gfx1250, wave32 + WMMA.
//   B=64, N=577, E=768, H=12, D=DV=64, R=403 (R1=404)
// All four matmul stages run through v_wmma_f32_16x16x32_bf16.
// LDS tiles live in WMMA fragment order (contiguous 32B per lane ->
// ds_load_b128 pairs), staging loads are batched into clauses, and the two
// K-loop kernels double-buffer LDS so staging overlaps the WMMA stream.
// ---------------------------------------------------------------------------

#define B_   64
#define N_   577
#define E_   768
#define H_   12
#define Dh   64
#define BH_  (B_ * H_)
#define R_   403
#define R1_  404
#define PI_F 3.14159265358979323846f

typedef __attribute__((ext_vector_type(16))) __bf16 v16bf;
typedef __attribute__((ext_vector_type(8)))  __bf16 v8bf;
typedef __attribute__((ext_vector_type(8)))  float  v8f;

// A-fragment element mapping (16x32 bf16): K -> (half, j)
__device__ __forceinline__ int a_half(int k) { return (k >> 3) & 1; }
__device__ __forceinline__ int a_j(int k)    { return (k & 7) + ((k >> 4) << 3); }
// B-fragment element mapping (32x16 bf16): same K -> (half, j) as A

__device__ __forceinline__ float nonneg_f(float x) {
    return (x < 0.f) ? __expf(x) : (x + 1.f);
}

// ---------------------------------------------------------------------------
// Kernel 1: projection GEMM  Y = X @ W + b, written head-split as bf16:
//   Yh[(b*H + h)*N*64 + n*64 + d]  where row r=b*N+n, col c=h*64+d.
// grid (577, 12) ; block 128 (4 waves) ; tile 64(M) x 64(N), K-step 32,
// LDS double-buffered.
// ---------------------------------------------------------------------------
__device__ __forceinline__ void proj_stage(const float* __restrict__ X,
                                           const float* __restrict__ W,
                                           int mBlk, int nBlk, int k0, int tid,
                                           __bf16* __restrict__ As,
                                           __bf16* __restrict__ Bs)
{
    float2 xa[8], xb[8];
    #pragma unroll
    for (int t = 0; t < 8; ++t) {               // A: 64 rows x 16 pairs
        int pi = tid + t * 128;
        int r  = pi >> 4;
        int kp = (pi & 15) << 1;
        xa[t] = *(const float2*)&X[(size_t)(mBlk + r) * E_ + k0 + kp];
    }
    #pragma unroll
    for (int t = 0; t < 8; ++t) {               // B: 32 rows x 32 pairs
        int pi = tid + t * 128;
        int r  = pi >> 5;
        int cp = (pi & 31) << 1;
        xb[t] = *(const float2*)&W[(size_t)(k0 + r) * E_ + nBlk + cp];
    }
    #pragma unroll
    for (int t = 0; t < 8; ++t) {
        int pi = tid + t * 128;
        int r  = pi >> 4;
        int kp = (pi & 15) << 1;
        __bf16* dst = &As[((r * 2 + a_half(kp)) << 4) + a_j(kp)];
        dst[0] = (__bf16)xa[t].x;
        dst[1] = (__bf16)xa[t].y;
    }
    #pragma unroll
    for (int t = 0; t < 8; ++t) {
        int pi = tid + t * 128;
        int r  = pi >> 5;
        int cp = (pi & 31) << 1;
        int hb = a_half(r), jb = a_j(r);
        Bs[((cp * 2 + hb) << 4) + jb]       = (__bf16)xb[t].x;
        Bs[(((cp + 1) * 2 + hb) << 4) + jb] = (__bf16)xb[t].y;
    }
}

__global__ __launch_bounds__(128)
void proj_bf16_kernel(const float* __restrict__ X, const float* __restrict__ W,
                      const float* __restrict__ bias, __bf16* __restrict__ Yh)
{
    __shared__ __align__(32) __bf16 As[2][64 * 32];
    __shared__ __align__(32) __bf16 Bs[2][64 * 32];

    const int tid  = threadIdx.x;
    const int lane = tid & 31;
    const int wave = tid >> 5;
    const int half = lane >> 4;
    const int mBlk = blockIdx.x * 64;
    const int nBlk = blockIdx.y * 64;

    v8f acc[4] = {v8f{}, v8f{}, v8f{}, v8f{}};

    proj_stage(X, W, mBlk, nBlk, 0, tid, As[0], Bs[0]);
    __syncthreads();

    int buf = 0;
    const int m = lane & 15, ncol = lane & 15;
    for (int k0 = 0; k0 < E_; k0 += 32) {
        if (k0 + 32 < E_) {
            __builtin_prefetch(&X[(size_t)(mBlk + (tid >> 1)) * E_ + k0 + 64], 0, 1);
            proj_stage(X, W, mBlk, nBlk, k0 + 32, tid, As[buf ^ 1], Bs[buf ^ 1]);
        }
        v16bf a = *reinterpret_cast<const v16bf*>(
            &As[buf][(((wave * 16 + m) * 2 + half) << 4)]);
        #pragma unroll
        for (int nt = 0; nt < 4; ++nt) {
            v16bf bfr = *reinterpret_cast<const v16bf*>(
                &Bs[buf][(((nt * 16 + ncol) * 2 + half) << 4)]);
            acc[nt] = __builtin_amdgcn_wmma_f32_16x16x32_bf16(
                false, a, false, bfr, (short)0, acc[nt], false, false);
        }
        __syncthreads();
        buf ^= 1;
    }

    // store head-split bf16 (+bias)
    const int rb = half * 8;
    for (int nt = 0; nt < 4; ++nt) {
        int c = nBlk + nt * 16 + ncol;
        int h = c >> 6, d = c & 63;
        float bv = bias[c];
        #pragma unroll
        for (int r = 0; r < 8; ++r) {
            int row = mBlk + wave * 16 + rb + r;
            int bidx = row / N_;
            int n    = row - bidx * N_;
            Yh[((size_t)(bidx * H_ + h) * N_ + n) * Dh + d] = (__bf16)(acc[nt][r] + bv);
        }
    }
}

// ---------------------------------------------------------------------------
// Kernel 2: DistributionSample. One block per (b,h): logits = q0.k[1:]/8,
// log_softmax, +gumbel(hash), rank-select top R_, emit sorted sel[R1_].
// grid 768 ; block 576.
// ---------------------------------------------------------------------------
__global__ __launch_bounds__(576)
void sample_kernel(const __bf16* __restrict__ QH, const __bf16* __restrict__ KH,
                   int* __restrict__ sel)
{
    const int bh = blockIdx.x;
    const int j  = threadIdx.x;            // 0..575
    __shared__ float keys[576];
    __shared__ int   mask[576];
    __shared__ float red[2];

    const v8bf* q0 = (const v8bf*)(QH + (size_t)bh * N_ * Dh);          // token 0
    const v8bf* kr = (const v8bf*)(KH + ((size_t)bh * N_ + 1 + j) * Dh);
    float dot = 0.f;
    #pragma unroll
    for (int t = 0; t < 8; ++t) {
        v8bf a = q0[t], b = kr[t];
        #pragma unroll
        for (int e = 0; e < 8; ++e) dot += (float)a[e] * (float)b[e];
    }
    float logit = dot * 0.125f;            // 1/sqrt(64)
    keys[j] = logit;
    __syncthreads();

    if (j == 0) {
        float mx = -1e30f;
        for (int i = 0; i < 576; ++i) mx = fmaxf(mx, keys[i]);
        float s = 0.f;
        for (int i = 0; i < 576; ++i) s += __expf(keys[i] - mx);
        red[0] = mx; red[1] = __logf(s);
    }
    __syncthreads();
    float logp = logit - red[0] - red[1];

    // deterministic gumbel stand-in (seed 42)
    unsigned int x = (unsigned)(bh * 576 + j) * 0x9E3779B9u + 42u * 0x85EBCA6Bu;
    x ^= x >> 16; x *= 0x7FEB352Du; x ^= x >> 15; x *= 0x846CA68Bu; x ^= x >> 16;
    float u = ((float)(x >> 8) + 0.5f) * (1.0f / 16777216.0f);
    float g = -__logf(-__logf(u));
    float key = logp + g;
    keys[j] = key;
    __syncthreads();

    int cnt = 0;
    for (int i = 0; i < 576; ++i) {
        float ki = keys[i];
        cnt += (ki > key) || (ki == key && i < j);
    }
    int selected = (cnt < R_) ? 1 : 0;
    mask[j] = selected;
    __syncthreads();

    if (selected) {
        int pos = 0;
        for (int i = 0; i < j; ++i) pos += mask[i];
        sel[(size_t)bh * R1_ + 1 + pos] = j + 1;   // tokens sorted ascending
    }
    if (j == 0) sel[(size_t)bh * R1_] = 0;         // CLS always kept
}

// ---------------------------------------------------------------------------
// Kernel 3: per-head accumulators  kcv = k_cos^T @ v, ksv = k_sin^T @ v.
// K/V slabs loaded as v8bf clauses, feature map fused on stage, LDS
// double-buffered.  grid 768 ; block 128 (4 waves).
// ---------------------------------------------------------------------------
__device__ __forceinline__ void kv_stage(const __bf16* __restrict__ kb,
                                         const __bf16* __restrict__ vb,
                                         const float* __restrict__ cosT,
                                         const float* __restrict__ sinT,
                                         int n0, int tid,
                                         __bf16* __restrict__ AcS,
                                         __bf16* __restrict__ AsS,
                                         __bf16* __restrict__ VbS)
{
    const v8bf zz = v8bf{};
    v8bf k8[2], v8[2];
    #pragma unroll
    for (int c = 0; c < 2; ++c) {              // 2048 elems = 256 v8bf chunks
        int base = (tid + c * 128) * 8;
        int nl = base >> 6, d0 = base & 63;
        int n  = n0 + nl;
        bool ok = n < N_;
        k8[c] = ok ? *(const v8bf*)&kb[(size_t)n * Dh + d0] : zz;
        v8[c] = ok ? *(const v8bf*)&vb[(size_t)n * Dh + d0] : zz;
    }
    #pragma unroll
    for (int c = 0; c < 2; ++c) {
        int base = (tid + c * 128) * 8;
        int nl = base >> 6, d0 = base & 63;
        int n  = n0 + nl;
        bool ok = n < N_;
        float cn = cosT[n], sn = sinT[n];
        int ah = a_half(nl), aj = a_j(nl);
        int bh2 = a_half(nl), bj = a_j(nl);
        #pragma unroll
        for (int e = 0; e < 8; ++e) {
            int d = d0 + e;
            float f = ok ? nonneg_f((float)k8[c][e]) : 0.f;
            int ai = ((d * 2 + ah) << 4) + aj;
            AcS[ai] = (__bf16)(f * cn);
            AsS[ai] = (__bf16)(f * sn);
            VbS[((d * 2 + bh2) << 4) + bj] = v8[c][e];
        }
    }
}

__global__ __launch_bounds__(128)
void kv_accum_kernel(const __bf16* __restrict__ KH, const __bf16* __restrict__ VH,
                     float* __restrict__ KCV, float* __restrict__ KSV)
{
    __shared__ __align__(32) __bf16 AcS[2][64 * 32];
    __shared__ __align__(32) __bf16 AsS[2][64 * 32];
    __shared__ __align__(32) __bf16 VbS[2][64 * 32];
    __shared__ float cosT[608], sinT[608];

    const int bh   = blockIdx.x;
    const int tid  = threadIdx.x;
    const int lane = tid & 31;
    const int wave = tid >> 5;
    const int half = lane >> 4;
    const __bf16* kb = KH + (size_t)bh * N_ * Dh;
    const __bf16* vb = VH + (size_t)bh * N_ * Dh;

    const float PSC = PI_F * 0.5f / (float)N_;
    for (int n = tid; n < 608; n += 128) {
        float p = (float)n * PSC;
        cosT[n] = __cosf(p);
        sinT[n] = __sinf(p);
    }
    __syncthreads();

    kv_stage(kb, vb, cosT, sinT, 0, tid, AcS[0], AsS[0], VbS[0]);
    __syncthreads();

    v8f accC[4] = {v8f{}, v8f{}, v8f{}, v8f{}};
    v8f accS[4] = {v8f{}, v8f{}, v8f{}, v8f{}};
    const int m = lane & 15, ncol = lane & 15;

    int buf = 0;
    for (int n0 = 0; n0 < N_; n0 += 32) {
        if (n0 + 32 < N_)
            kv_stage(kb, vb, cosT, sinT, n0 + 32, tid,
                     AcS[buf ^ 1], AsS[buf ^ 1], VbS[buf ^ 1]);

        int arow = ((wave * 16 + m) * 2 + half) << 4;
        v16bf ac = *reinterpret_cast<const v16bf*>(&AcS[buf][arow]);
        v16bf as = *reinterpret_cast<const v16bf*>(&AsS[buf][arow]);
        #pragma unroll
        for (int nt = 0; nt < 4; ++nt) {
            v16bf bfr = *reinterpret_cast<const v16bf*>(
                &VbS[buf][(((nt * 16 + ncol) * 2 + half) << 4)]);
            accC[nt] = __builtin_amdgcn_wmma_f32_16x16x32_bf16(
                false, ac, false, bfr, (short)0, accC[nt], false, false);
            accS[nt] = __builtin_amdgcn_wmma_f32_16x16x32_bf16(
                false, as, false, bfr, (short)0, accS[nt], false, false);
        }
        __syncthreads();
        buf ^= 1;
    }

    const int rb = half * 8;
    for (int nt = 0; nt < 4; ++nt) {
        int dv = nt * 16 + ncol;
        #pragma unroll
        for (int r = 0; r < 8; ++r) {
            int d = wave * 16 + rb + r;
            KCV[((size_t)bh * 64 + d) * 64 + dv] = accC[nt][r];
            KSV[((size_t)bh * 64 + d) * 64 + dv] = accS[nt][r];
        }
    }
}

// ---------------------------------------------------------------------------
// Kernel 4: sampled = q_cos_s @ kcv + q_sin_s @ ksv, scattered into out.
// grid (7, 768) ; block 128 (4 waves, wave w owns m-tile blockIdx.x*4+w).
// kcv/ksv staged bf16-transposed via float4 clauses; Q rows via v8bf loads.
// ---------------------------------------------------------------------------
__global__ __launch_bounds__(128)
void out_gemm_scatter_kernel(const __bf16* __restrict__ QH,
                             const float* __restrict__ KCV,
                             const float* __restrict__ KSV,
                             const int* __restrict__ sel,
                             float* __restrict__ out)
{
    __shared__ __align__(32) __bf16 BcS[64 * 64];  // [dv*64 + kk] == B frag order
    __shared__ __align__(32) __bf16 BsS[64 * 64];

    const int bh   = blockIdx.y;
    const int b    = bh / H_;
    const int h    = bh - b * H_;
    const int tid  = threadIdx.x;
    const int lane = tid & 31;
    const int wave = tid >> 5;
    const int half = lane >> 4;

    const int*    sb  = sel + (size_t)bh * R1_;
    const __bf16* qb  = QH  + (size_t)bh * N_ * Dh;
    const float*  kcv = KCV + (size_t)bh * 4096;
    const float*  ksv = KSV + (size_t)bh * 4096;

    // stage transposed bf16 copies of kcv/ksv: batched float4 clauses
    {
        float4 xc[8], xs[8];
        #pragma unroll
        for (int t = 0; t < 8; ++t) {
            int base = (tid + t * 128) * 4;
            xc[t] = *(const float4*)&kcv[base];
            xs[t] = *(const float4*)&ksv[base];
        }
        #pragma unroll
        for (int t = 0; t < 8; ++t) {
            int base = (tid + t * 128) * 4;
            int kk = base >> 6, dv0 = base & 63;
            BcS[(dv0 + 0) * 64 + kk] = (__bf16)xc[t].x;
            BcS[(dv0 + 1) * 64 + kk] = (__bf16)xc[t].y;
            BcS[(dv0 + 2) * 64 + kk] = (__bf16)xc[t].z;
            BcS[(dv0 + 3) * 64 + kk] = (__bf16)xc[t].w;
            BsS[(dv0 + 0) * 64 + kk] = (__bf16)xs[t].x;
            BsS[(dv0 + 1) * 64 + kk] = (__bf16)xs[t].y;
            BsS[(dv0 + 2) * 64 + kk] = (__bf16)xs[t].z;
            BsS[(dv0 + 3) * 64 + kk] = (__bf16)xs[t].w;
        }
    }
    __syncthreads();

    const int mtile = blockIdx.x * 4 + wave;      // 0..27, need 0..25
    if (mtile >= (R1_ + 15) / 16) return;

    v8f acc[4] = {v8f{}, v8f{}, v8f{}, v8f{}};
    const int m    = lane & 15;
    const int ncol = lane & 15;
    const int kb8  = half * 8;

    const int  sr    = mtile * 16 + m;
    const bool valid = sr < R1_;
    const int  tok   = valid ? sb[sr] : 0;
    const float PSC  = PI_F * 0.5f / (float)N_;
    const float p    = (float)tok * PSC;
    const float cp   = __cosf(p), sp = __sinf(p);

    const v8bf* qrow = (const v8bf*)(qb + (size_t)tok * Dh);
    const v8bf  zz   = v8bf{};

    #pragma unroll
    for (int ki = 0; ki < 2; ++ki) {       // K = 0..31, 32..63
        v8bf lo = valid ? qrow[(ki * 32 + kb8 * 2) >> 3]     : zz;
        v8bf hi = valid ? qrow[(ki * 32 + kb8 * 2 + 8) >> 3] : zz;
        v16bf aC, aS;
        #pragma unroll
        for (int t = 0; t < 8; ++t) {
            float f0 = valid ? nonneg_f((float)lo[t]) : 0.f;
            float f1 = valid ? nonneg_f((float)hi[t]) : 0.f;
            aC[t]     = (__bf16)(f0 * cp);
            aS[t]     = (__bf16)(f0 * sp);
            aC[t + 8] = (__bf16)(f1 * cp);
            aS[t + 8] = (__bf16)(f1 * sp);
        }
        #pragma unroll
        for (int nt = 0; nt < 4; ++nt) {
            int dv = nt * 16 + ncol;
            v16bf bC = *reinterpret_cast<const v16bf*>(&BcS[dv * 64 + ki * 32 + half * 16]);
            v16bf bS = *reinterpret_cast<const v16bf*>(&BsS[dv * 64 + ki * 32 + half * 16]);
            acc[nt] = __builtin_amdgcn_wmma_f32_16x16x32_bf16(
                false, aC, false, bC, (short)0, acc[nt], false, false);
            acc[nt] = __builtin_amdgcn_wmma_f32_16x16x32_bf16(
                false, aS, false, bS, (short)0, acc[nt], false, false);
        }
    }

    // scatter rows back to token positions (unique tokens -> plain stores)
    const int rb = half * 8;
    for (int nt = 0; nt < 4; ++nt) {
        int dv = nt * 16 + ncol;
        #pragma unroll
        for (int r = 0; r < 8; ++r) {
            int srr = mtile * 16 + rb + r;
            if (srr < R1_) {
                int t = sb[srr];
                out[((size_t)b * N_ + t) * E_ + h * Dh + dv] = acc[nt][r];
            }
        }
    }
}

// ---------------------------------------------------------------------------
__global__ __launch_bounds__(256)
void zero_f32_kernel(float* __restrict__ p, size_t n)
{
    size_t i      = (size_t)blockIdx.x * blockDim.x + threadIdx.x;
    size_t stride = (size_t)gridDim.x * blockDim.x;
    for (; i < n; i += stride) p[i] = 0.f;
}

// ---------------------------------------------------------------------------
extern "C" void kernel_launch(void* const* d_in, const int* in_sizes, int n_in,
                              void* d_out, int out_size, void* d_ws, size_t ws_size,
                              hipStream_t stream)
{
    (void)in_sizes; (void)n_in; (void)out_size; (void)ws_size;
    const float* q  = (const float*)d_in[0];
    const float* k  = (const float*)d_in[1];
    const float* v  = (const float*)d_in[2];
    const float* Wq = (const float*)d_in[3];
    const float* Wk = (const float*)d_in[4];
    const float* Wv = (const float*)d_in[5];
    const float* bq = (const float*)d_in[6];
    const float* bk = (const float*)d_in[7];
    const float* bv = (const float*)d_in[8];
    float* out = (float*)d_out;

    // workspace carve-out (256B aligned)
    char*  ws  = (char*)d_ws;
    size_t off = 0;
    auto carve = [&](size_t bytes) -> void* {
        void* p = ws + off;
        off += (bytes + 255) & ~(size_t)255;
        return p;
    };
    const size_t headElems = (size_t)BH_ * N_ * Dh;           // 28,360,704
    __bf16* QH  = (__bf16*)carve(headElems * sizeof(__bf16));
    __bf16* KH  = (__bf16*)carve(headElems * sizeof(__bf16));
    __bf16* VH  = (__bf16*)carve(headElems * sizeof(__bf16));
    float*  KCV = (float*) carve((size_t)BH_ * 4096 * sizeof(float));
    float*  KSV = (float*) carve((size_t)BH_ * 4096 * sizeof(float));
    int*    SEL = (int*)   carve((size_t)BH_ * R1_ * sizeof(int));

    // 1) projections -> head-split bf16   (M = B*N = 577*64 tiles exactly)
    dim3 gProj(577, E_ / 64);
    proj_bf16_kernel<<<gProj, 128, 0, stream>>>(q, Wq, bq, QH);
    proj_bf16_kernel<<<gProj, 128, 0, stream>>>(k, Wk, bk, KH);
    proj_bf16_kernel<<<gProj, 128, 0, stream>>>(v, Wv, bv, VH);

    // 2) token sampling per head
    sample_kernel<<<BH_, 576, 0, stream>>>(QH, KH, SEL);

    // 3) k^T v accumulators per head
    kv_accum_kernel<<<BH_, 128, 0, stream>>>(KH, VH, KCV, KSV);

    // 4) zero output, then gathered GEMM + scatter
    zero_f32_kernel<<<2048, 256, 0, stream>>>(out, (size_t)B_ * N_ * E_);
    dim3 gOut(((R1_ + 15) / 16 + 3) / 4, BH_);
    out_gemm_scatter_kernel<<<gOut, 128, 0, stream>>>(QH, KCV, KSV, SEL, out);
}
